// CoAttention_64982855188557
// MI455X (gfx1250) — matrix operand order repaired and checked by reading, not compiled, not run on hardware
//
#include <hip/hip_runtime.h>
#include <math.h>

typedef __attribute__((ext_vector_type(16))) __bf16 v16bf;
typedef __attribute__((ext_vector_type(8)))  float  v8f;

// Problem dims
#define B_   16
#define S_   1024
#define R_   512
#define H_   512
#define HD_  256
#define G_   1024
#define DIN_ 1536

__device__ __forceinline__ float sigm(float x) { return 1.0f / (1.0f + expf(-x)); }

// Compile-time-stride pair load (K and K+1)
template<long STRIDE>
__device__ __forceinline__ float2 ld2c(const float* p) {
  if constexpr (STRIDE == 1) {
    return *reinterpret_cast<const float2*>(p);
  } else {
    float2 r; r.x = p[0]; r.y = p[STRIDE]; return r;
  }
}

__device__ __forceinline__ v16bf pack_bf16(const float2* r) {
  v16bf f;
#pragma unroll
  for (int v = 0; v < 8; ++v) {
    f[2 * v]     = (__bf16)r[v].x;
    f[2 * v + 1] = (__bf16)r[v].y;
  }
  return f;
}

// ---------------------------------------------------------------------------
// Strided batched GEMM via bf16 WMMA.  One wave per 32x32 C tile (2x2 of
// 16x16).  All strides/K compile-time -> immediate-offset loads, minimal VALU.
//   C[m][n] = act( sum_k A[m][k]*Bn[n][k] + bias0[n] + bias1[n] )
// ---------------------------------------------------------------------------
template<int KTOT, long LDA, long SKA, long LDB, long SKB, long LDC, int ACT>
__global__ __launch_bounds__(32) void gemm_tile(
    const float* __restrict__ A, long batchA,
    const float* __restrict__ Bm, long batchB,
    float* __restrict__ C, long batchC,
    const float* __restrict__ bias0, const float* __restrict__ bias1)
{
  const int lane = threadIdx.x & 31;
  const int hi   = lane >> 4;      // lane half
  const int ln   = lane & 15;
  const long nBase = (long)blockIdx.x * 32;
  const long mBase = (long)blockIdx.y * 32;

  // Fold lane-half K offset into base pointers:
  //  A frag: ka = hi*8  + (v<4 ? 2v : 2v+8)
  //  B frag: kb = hi*8  + (v<4 ? 2v : 2v+8)
  const float* Ah0 = A + (long)blockIdx.z * batchA + (mBase + ln) * LDA + (hi ? 8 : 0) * SKA;
  const float* Ah1 = Ah0 + 16 * LDA;
  const float* Bh0 = Bm + (long)blockIdx.z * batchB + (nBase + ln) * LDB + (hi ? 8 : 0) * SKB;
  const float* Bh1 = Bh0 + 16 * LDB;
  float* Cb = C + (long)blockIdx.z * batchC;

  v8f acc00 = {}, acc01 = {}, acc10 = {}, acc11 = {};

#pragma unroll 2
  for (int kk = 0; kk < KTOT; kk += 32) {
    float2 ra0[8], ra1[8], rb0[8], rb1[8];
#pragma unroll
    for (int v = 0; v < 8; ++v) {
      const long ka = (v < 4) ? (2 * v) : (2 * v + 8);   // compile-time
      const long kb = (v < 4) ? (2 * v) : (2 * v + 8);   // compile-time
      ra0[v] = ld2c<SKA>(Ah0 + ka * SKA);
      ra1[v] = ld2c<SKA>(Ah1 + ka * SKA);
      rb0[v] = ld2c<SKB>(Bh0 + kb * SKB);
      rb1[v] = ld2c<SKB>(Bh1 + kb * SKB);
    }
    Ah0 += 32 * SKA; Ah1 += 32 * SKA;
    Bh0 += 32 * SKB; Bh1 += 32 * SKB;

    v16bf a0 = pack_bf16(ra0), a1 = pack_bf16(ra1);
    v16bf b0 = pack_bf16(rb0), b1 = pack_bf16(rb1);

    acc00 = __builtin_amdgcn_wmma_f32_16x16x32_bf16(false, a0, false, b0, (short)0, acc00, false, false);
    acc01 = __builtin_amdgcn_wmma_f32_16x16x32_bf16(false, a0, false, b1, (short)0, acc01, false, false);
    acc10 = __builtin_amdgcn_wmma_f32_16x16x32_bf16(false, a1, false, b0, (short)0, acc10, false, false);
    acc11 = __builtin_amdgcn_wmma_f32_16x16x32_bf16(false, a1, false, b1, (short)0, acc11, false, false);
  }

  float bv0 = 0.0f, bv1 = 0.0f;
  if (bias0) { bv0 += bias0[nBase + ln]; bv1 += bias0[nBase + 16 + ln]; }
  if (bias1) { bv0 += bias1[nBase + ln]; bv1 += bias1[nBase + 16 + ln]; }

  // f32 C/D layout: VGPR r, lanes 0-15 -> M=r, lanes 16-31 -> M=r+8; N=ln
  float* Crow0 = Cb + (mBase + hi * 8) * LDC + nBase + ln;
  float* Crow1 = Crow0 + 16 * LDC;
#pragma unroll
  for (int r = 0; r < 8; ++r) {
    float x00 = acc00[r] + bv0, x01 = acc01[r] + bv1;
    float x10 = acc10[r] + bv0, x11 = acc11[r] + bv1;
    if (ACT == 1) { x00 = tanhf(x00); x01 = tanhf(x01); x10 = tanhf(x10); x11 = tanhf(x11); }
    Crow0[(long)r * LDC]      = x00;
    Crow0[(long)r * LDC + 16] = x01;
    Crow1[(long)r * LDC]      = x10;
    Crow1[(long)r * LDC + 16] = x11;
  }
}

// ---------------------------------------------------------------------------
// Generic line softmax: one 256-thread block per line of n elements.
// ---------------------------------------------------------------------------
__global__ __launch_bounds__(256) void softmax_lines(
    const float* __restrict__ in, float* __restrict__ outp,
    int n, long estride, long lstride, int lines_per_batch, long bstride)
{
  long line = blockIdx.x;
  long base = (line / lines_per_batch) * bstride + (line % lines_per_batch) * lstride;
  const float* ip = in + base;
  float* op = outp + base;
  __shared__ float red[256];

  float mx = -3.4e38f;
  for (int i = threadIdx.x; i < n; i += 256) mx = fmaxf(mx, ip[(long)i * estride]);
  red[threadIdx.x] = mx; __syncthreads();
  for (int o = 128; o > 0; o >>= 1) {
    if ((int)threadIdx.x < o) red[threadIdx.x] = fmaxf(red[threadIdx.x], red[threadIdx.x + o]);
    __syncthreads();
  }
  mx = red[0]; __syncthreads();

  float s = 0.0f;
  for (int i = threadIdx.x; i < n; i += 256) s += expf(ip[(long)i * estride] - mx);
  red[threadIdx.x] = s; __syncthreads();
  for (int o = 128; o > 0; o >>= 1) {
    if ((int)threadIdx.x < o) red[threadIdx.x] += red[threadIdx.x + o];
    __syncthreads();
  }
  float inv = 1.0f / red[0];
  for (int i = threadIdx.x; i < n; i += 256)
    op[(long)i * estride] = expf(ip[(long)i * estride] - mx) * inv;
}

// x[r][b][2H + h] = ref[r][b][h]
__global__ void copy_href(const float* __restrict__ ref, float* __restrict__ x)
{
  long i = (long)blockIdx.x * blockDim.x + threadIdx.x;  // over R*B*H
  if (i >= (long)R_ * B_ * H_) return;
  long h = i % H_;
  long rb = i / H_;
  long b = rb % B_;
  long r = rb / B_;
  x[(r * B_ + b) * (long)DIN_ + 2 * H_ + h] = ref[i];
}

// ---------------------------------------------------------------------------
// LSTM scan: one 1024-thread workgroup per direction (blockIdx.x = dir).
// 32 waves = 16 tile-pairs x 2 K-halves.  Wave (p, kh) owns gate tiles
// {2p, 2p+1} over K in [kh*128, kh*128+128): bw = 2x4 frags = 64 VGPRs ->
// fits the 128-VGPR cap at 8 waves/SIMD, no scratch spills.
// K-high waves store bf16 partials to gates_sh; K-low waves (which folded in
// xg) add the partial and overwrite the slot with final gates; owner threads
// then do the c/h update with c in registers.  3 barriers/step.
// ---------------------------------------------------------------------------
__global__ __launch_bounds__(1024) void lstm_scan(
    const float* __restrict__ xg_f, const float* __restrict__ xg_b,
    const float* __restrict__ Whh_f, const float* __restrict__ Whh_b,
    float* __restrict__ out)
{
  const int dir = blockIdx.x;
  const float* xg  = dir ? xg_b  : xg_f;
  const float* Whh = dir ? Whh_b : Whh_f;

  const int tid   = threadIdx.x;
  const int lane  = tid & 31;
  const int wave  = tid >> 5;     // 0..31
  const int pairw = wave & 15;    // tile-pair index
  const int kh    = wave >> 4;    // K half (0: K 0..127 + xg, 1: K 128..255)
  const int hi    = lane >> 4;
  const int ln    = lane & 15;

  __shared__ __bf16 gates_sh[G_][B_];       // 32 KB: partials, then final gates
  __shared__ __bf16 h_sh[B_][HD_ + 8];      // 8.4 KB, A-source layout

  // Preload W_hh B-fragments: 2 tiles x 4 K-chunks = 64 VGPRs
  v16bf bw[2][4];
#pragma unroll
  for (int gt = 0; gt < 2; ++gt) {
    const int nTile = 2 * pairw + gt;
    const float* wb = Whh + (long)(nTile * 16 + ln) * HD_ + kh * 128 + (hi ? 8 : 0);
#pragma unroll
    for (int j = 0; j < 4; ++j) {
      float2 rw[8];
#pragma unroll
      for (int v = 0; v < 8; ++v)
        rw[v] = *reinterpret_cast<const float2*>(wb + j * 32 + ((v < 4) ? (2 * v) : (2 * v + 8)));
      bw[gt][j] = pack_bf16(rw);
    }
  }

  for (int i = tid; i < B_ * (HD_ + 8); i += 1024)
    ((__bf16*)h_sh)[i] = (__bf16)0.0f;

  // Elementwise ownership: thread owns (b = tid>>6, hd = (tid&63)*4 .. +3)
  const int ob  = tid >> 6;
  const int ohd = (tid & 63) * 4;
  float creg[4];
#pragma unroll
  for (int e = 0; e < 4; ++e) creg[e] = 0.0f;
  __syncthreads();

  // Per-step base pointers with lane terms folded in; advance by +/- stride.
  const long t0   = dir ? (R_ - 1) : 0;
  const long dxg  = dir ? -(long)B_ * G_ : (long)B_ * G_;
  const long dout = dir ? -(long)B_ * 2 * HD_ : (long)B_ * 2 * HD_;
  const float* xptr = xg + t0 * B_ * G_ + (long)hi * 8 * G_ + 32 * pairw + ln;
  float* optr = out + t0 * B_ * 2 * HD_ + (long)ob * 2 * HD_ + (long)(dir * HD_) + ohd;
  const __bf16* hrow = &h_sh[ln][kh * 128 + (hi ? 8 : 0)];  // + j*32 + (v<4 ? 2v : 2v+8)
  __bf16* gs0 = &gates_sh[(2 * pairw + 0) * 16 + ln][hi * 8];
  __bf16* gs1 = &gates_sh[(2 * pairw + 1) * 16 + ln][hi * 8];
  const __bf16* gld = &gates_sh[ohd][ob];   // + (gate*256 + e)*B_

  for (int step = 0; step < R_; ++step) {
    // K-low waves fold xg into the accumulator; K-high waves start at zero.
    v8f acc0 = {}, acc1 = {};
    if (kh == 0) {
#pragma unroll
      for (int r = 0; r < 8; ++r) {
        acc0[r] = xptr[(long)r * G_];
        acc1[r] = xptr[(long)r * G_ + 16];
      }
    }

    // 4 K-chunks: A frag from LDS (paired ds_load_b128) feeds both tiles
#pragma unroll
    for (int j = 0; j < 4; ++j) {
      v16bf af;
#pragma unroll
      for (int v = 0; v < 8; ++v) {
        const int ka = j * 32 + ((v < 4) ? (2 * v) : (2 * v + 8));
        af[2 * v]     = hrow[ka];
        af[2 * v + 1] = hrow[ka + 1];
      }
      acc0 = __builtin_amdgcn_wmma_f32_16x16x32_bf16(false, af, false, bw[0][j], (short)0, acc0, false, false);
      acc1 = __builtin_amdgcn_wmma_f32_16x16x32_bf16(false, af, false, bw[1][j], (short)0, acc1, false, false);
    }

    // K-high waves publish bf16 partials (one 16B LDS store per tile)
    if (kh == 1) {
      __attribute__((aligned(16))) __bf16 p0[8], p1[8];
#pragma unroll
      for (int r = 0; r < 8; ++r) { p0[r] = (__bf16)acc0[r]; p1[r] = (__bf16)acc1[r]; }
      *reinterpret_cast<float4*>(gs0) = *reinterpret_cast<const float4*>(p0);
      *reinterpret_cast<float4*>(gs1) = *reinterpret_cast<const float4*>(p1);
    }
    __syncthreads();   // partials visible; previous h_sh reads complete

    // K-low waves combine and overwrite with final gates
    if (kh == 0) {
      __attribute__((aligned(16))) __bf16 p0[8], p1[8];
      *reinterpret_cast<float4*>(p0) = *reinterpret_cast<const float4*>(gs0);
      *reinterpret_cast<float4*>(p1) = *reinterpret_cast<const float4*>(gs1);
#pragma unroll
      for (int r = 0; r < 8; ++r) {
        p0[r] = (__bf16)(acc0[r] + (float)p0[r]);
        p1[r] = (__bf16)(acc1[r] + (float)p1[r]);
      }
      *reinterpret_cast<float4*>(gs0) = *reinterpret_cast<const float4*>(p0);
      *reinterpret_cast<float4*>(gs1) = *reinterpret_cast<const float4*>(p1);
    }
    __syncthreads();   // final gates visible

    // Elementwise c/h update on owned cells
    float hv[4];
#pragma unroll
    for (int e = 0; e < 4; ++e) {
      float ig = (float)gld[(0 * 256 + e) * B_];
      float fg = (float)gld[(1 * 256 + e) * B_];
      float gg = (float)gld[(2 * 256 + e) * B_];
      float og = (float)gld[(3 * 256 + e) * B_];
      float c  = sigm(fg) * creg[e] + sigm(ig) * tanhf(gg);
      creg[e]  = c;
      hv[e]    = sigm(og) * tanhf(c);
      h_sh[ob][ohd + e] = (__bf16)hv[e];
    }
    *reinterpret_cast<float4*>(optr) = *reinterpret_cast<const float4*>(hv);

    __syncthreads();   // h_sh ready; gates_sh reads complete

    xptr += dxg;
    optr += dout;
  }
}

// ---------------------------------------------------------------------------
extern "C" void kernel_launch(void* const* d_in, const int* in_sizes, int n_in,
                              void* d_out, int out_size, void* d_ws, size_t ws_size,
                              hipStream_t stream) {
  (void)in_sizes; (void)n_in; (void)out_size; (void)ws_size;
  const float* src    = (const float*)d_in[1];   // [S][B][H]
  const float* refmb  = (const float*)d_in[2];   // [R][B][H]
  const float* W_ref  = (const float*)d_in[3];   // [H][H]
  const float* b_ref  = (const float*)d_in[4];   // [H]
  const float* W_ih_f = (const float*)d_in[5];   // [G][DIN]
  const float* W_hh_f = (const float*)d_in[6];   // [G][HD]
  const float* b_ih_f = (const float*)d_in[7];
  const float* b_hh_f = (const float*)d_in[8];
  const float* W_ih_b = (const float*)d_in[9];
  const float* W_hh_b = (const float*)d_in[10];
  const float* b_ih_b = (const float*)d_in[11];
  const float* b_hh_b = (const float*)d_in[12];

  float* ws = (float*)d_ws;
  float* hs   = ws;                                // [B][S][H]
  float* lbuf = hs   + (long)B_ * S_ * H_;         // [B][S][R]
  float* as_  = lbuf + (long)B_ * S_ * R_;         // [B][S][R]
  float* ar_  = as_  + (long)B_ * S_ * R_;         // [B][S][R]
  float* cs_  = ar_  + (long)B_ * S_ * R_;         // [B][S][H]  (c_s transposed)
  float* x    = cs_  + (long)B_ * S_ * H_;         // [R][B][DIN]
  float* xgf  = x    + (long)R_ * B_ * DIN_;       // [R][B][G]
  float* xgb  = xgf  + (long)R_ * B_ * G_;         // [R][B][G]

  // 1) h_s[b][s][o] = tanh(sum_h src[s][b][h]*W_ref[o][h] + b_ref[o])
  gemm_tile<H_, (long)B_ * H_, 1, H_, 1, H_, 1>
      <<<dim3(H_ / 32, S_ / 32, B_), 32, 0, stream>>>(
      src, (long)H_, W_ref, 0, hs, (long)S_ * H_, b_ref, nullptr);

  // 2) l[b][s][r] = sum_h hs[b][s][h] * ref[r][b][h]
  gemm_tile<H_, H_, 1, (long)B_ * H_, 1, R_, 0>
      <<<dim3(R_ / 32, S_ / 32, B_), 32, 0, stream>>>(
      hs, (long)S_ * H_, refmb, (long)H_, lbuf, (long)S_ * R_, nullptr, nullptr);

  // 3) softmaxes: a_s over s (strided), a_r over r (contiguous)
  softmax_lines<<<B_ * R_, 256, 0, stream>>>(lbuf, as_, S_, (long)R_, 1, R_, (long)S_ * R_);
  softmax_lines<<<B_ * S_, 256, 0, stream>>>(lbuf, ar_, R_, 1, (long)R_, S_, (long)S_ * R_);

  // 4) cs_[b][s][h] = sum_r as_[b][s][r] * ref[r][b][h]
  gemm_tile<R_, R_, 1, 1, (long)B_ * H_, H_, 0>
      <<<dim3(H_ / 32, S_ / 32, B_), 32, 0, stream>>>(
      as_, (long)S_ * R_, refmb, (long)H_, cs_, (long)S_ * H_, nullptr, nullptr);

  // 5) c_r into x[:, :, 0:2H]:  x[r][b][k] = sum_s cat[k][s] * ar_[b][s][r]
  gemm_tile<S_, 1, R_, 1, H_, (long)B_ * DIN_, 0>
      <<<dim3(H_ / 32, R_ / 32, B_), 32, 0, stream>>>(
      ar_, (long)S_ * R_, hs, (long)S_ * H_, x, (long)DIN_, nullptr, nullptr);
  gemm_tile<S_, 1, R_, 1, H_, (long)B_ * DIN_, 0>
      <<<dim3(H_ / 32, R_ / 32, B_), 32, 0, stream>>>(
      ar_, (long)S_ * R_, cs_, (long)S_ * H_, x + H_, (long)DIN_, nullptr, nullptr);

  // x[:, :, 2H:3H] = h_r
  copy_href<<<((long)R_ * B_ * H_ + 255) / 256, 256, 0, stream>>>(refmb, x);

  // 6) xg = x @ W_ih^T + b_ih + b_hh   (M = R*B flattened)
  gemm_tile<DIN_, DIN_, 1, DIN_, 1, G_, 0>
      <<<dim3(G_ / 32, (R_ * B_) / 32, 1), 32, 0, stream>>>(
      x, 0, W_ih_f, 0, xgf, 0, b_ih_f, b_hh_f);
  gemm_tile<DIN_, DIN_, 1, DIN_, 1, G_, 0>
      <<<dim3(G_ / 32, (R_ * B_) / 32, 1), 32, 0, stream>>>(
      x, 0, W_ih_b, 0, xgb, 0, b_ih_b, b_hh_b);

  // 7) sequential scan, one workgroup per direction
  lstm_scan<<<2, 1024, 0, stream>>>(xgf, xgb, W_hh_f, W_hh_b, (float*)d_out);
}
